// AttentionHiddenNet_21303037788804
// MI455X (gfx1250) — matrix operand order repaired and checked by reading, not compiled, not run on hardware
//
#include <hip/hip_runtime.h>

#define SEG 64
#define D_IN 256
#define D_OUT 128

// LDS strides (in elements), padded to break bank conflicts (row stride % 64 banks != 0)
#define AH_STR 264   // Ah  [64][256+8]  f16
#define WT_STR 264   // Wt  [128][256+8] f16 (W transposed: [n][k])
#define SN_STR 136   // segN[64][128+8]  f16 (row-major seg)
#define ST_STR 72    // segT[128][64+8]  f16 (seg transposed: [feat][token])
#define SC_STR 68    // scores[64][64+4] f32
#define AT_STR 72    // attn  [64][64+8] f16

typedef _Float16 half4_t __attribute__((ext_vector_type(4)));
typedef _Float16 half8_t __attribute__((ext_vector_type(8)));
typedef _Float16 v16h    __attribute__((ext_vector_type(16)));
typedef float    v8f     __attribute__((ext_vector_type(8)));

struct __align__(16) SmemPhase1 { _Float16 Ah[SEG * AH_STR]; };                 // 33792 B
struct __align__(16) SmemPhase2 { float scores[SEG * SC_STR];                   // 17408 B
                                  _Float16 attn[SEG * AT_STR]; };               //  9216 B
union  __align__(16) SmemUnion  { SmemPhase1 p1; SmemPhase2 p2; };
struct __align__(16) Smem {
  _Float16 Wt  [D_OUT * WT_STR];   // 67584 B
  _Float16 segN[SEG   * SN_STR];   // 17408 B
  _Float16 segT[D_OUT * ST_STR];   // 18432 B
  SmemUnion u;                     // 33792 B  -> total ~134 KB (fits 320 KB WGP LDS)
};

// A-fragment (16x32 f16, M-major source): lane holds K = k0+8h..+7 and +16..+23
__device__ __forceinline__ v16h ldA(const _Float16* base, int row0, int k0,
                                    int stride, int lane) {
  const int r = row0 + (lane & 15);
  const int c = k0 + ((lane >> 4) << 3);
  const _Float16* p = base + r * stride + c;
  half8_t lo = *reinterpret_cast<const half8_t*>(p);
  half8_t hi = *reinterpret_cast<const half8_t*>(p + 16);
  v16h out;
#pragma unroll
  for (int i = 0; i < 8; ++i) { out[i] = lo[i]; out[8 + i] = hi[i]; }
  return out;
}

// B-fragment (32x16 f16) loaded from the TRANSPOSED (N-major) buffer:
// lane holds K = k0+8h..+7 and +16..+23 for column N = n0 + (lane&15)
__device__ __forceinline__ v16h ldB(const _Float16* baseT, int n0, int k0,
                                    int stride, int lane) {
  const int r = n0 + (lane & 15);
  const int c = k0 + ((lane >> 4) << 3);
  const _Float16* p = baseT + r * stride + c;
  half8_t lo = *reinterpret_cast<const half8_t*>(p);
  half8_t hi = *reinterpret_cast<const half8_t*>(p + 16);
  v16h out;
#pragma unroll
  for (int i = 0; i < 8; ++i) { out[i] = lo[i]; out[8 + i] = hi[i]; }
  return out;
}

__device__ __forceinline__ v8f wmma16(v16h a, v16h b, v8f c) {
  return __builtin_amdgcn_wmma_f32_16x16x32_f16(false, a, false, b,
                                                (short)0, c, false, false);
}

__global__ __launch_bounds__(256)
void attn_hidden_fused(const float* __restrict__ h_states,
                       const float* __restrict__ W,
                       const float* __restrict__ bias,
                       const long long* __restrict__ seq_start_end,
                       float* __restrict__ out) {
  __shared__ Smem sm;
  const int t       = threadIdx.x;
  const int wave    = t >> 5;
  const int lane    = t & 31;
  const int half_id = lane >> 4;
  const int l16     = lane & 15;
  const int g       = blockIdx.x;

  const long long start = seq_start_end[2 * g];
  const float* hseg = h_states + (size_t)start * D_IN;

  // ---- Phase 0a: stage 64x256 f32 tile -> f16 Ah (contiguous 16384-float span)
#pragma unroll
  for (int j = 0; j < 16; ++j) {
    const int idx = j * 1024 + t * 4;
    const float4 v = *reinterpret_cast<const float4*>(hseg + idx);
    const int row = idx >> 8, col = idx & 255;
    half4_t hv = { (_Float16)v.x, (_Float16)v.y, (_Float16)v.z, (_Float16)v.w };
    *reinterpret_cast<half4_t*>(&sm.u.p1.Ah[row * AH_STR + col]) = hv;
  }
  // ---- Phase 0b: stage W (256x128 f32, L2-resident) -> f16 Wt transposed [n][k]
#pragma unroll
  for (int j = 0; j < 32; ++j) {
    const int idx = j * 1024 + t * 4;
    const float4 v = *reinterpret_cast<const float4*>(W + idx);
    const int k = idx >> 7, n = idx & 127;
    sm.Wt[(n + 0) * WT_STR + k] = (_Float16)v.x;
    sm.Wt[(n + 1) * WT_STR + k] = (_Float16)v.y;
    sm.Wt[(n + 2) * WT_STR + k] = (_Float16)v.z;
    sm.Wt[(n + 3) * WT_STR + k] = (_Float16)v.w;
  }
  __syncthreads();

  // ---- Phase 1: seg = Ah @ W + b  (64x128 in 16x16 tiles; wave -> 4 N-tiles)
  {
    const int m0 = (wave >> 1) * 16;
    v16h af[8];
#pragma unroll
    for (int ks = 0; ks < 8; ++ks)
      af[ks] = ldA(sm.u.p1.Ah, m0, ks * 32, AH_STR, lane);
#pragma unroll
    for (int nt = 0; nt < 4; ++nt) {
      const int n0 = ((wave & 1) * 4 + nt) * 16;
      v8f acc = {};
#pragma unroll
      for (int ks = 0; ks < 8; ++ks) {
        v16h bf = ldB(sm.Wt, n0, ks * 32, WT_STR, lane);
        acc = wmma16(af[ks], bf, acc);
      }
      const int col = n0 + l16;
      const float bv = bias[col];
#pragma unroll
      for (int v = 0; v < 8; ++v) {
        const int row = m0 + v + 8 * half_id;
        const _Float16 hx = (_Float16)(acc[v] + bv);
        sm.segN[row * SN_STR + col] = hx;
        sm.segT[col * ST_STR + row] = hx;
      }
    }
  }
  __syncthreads();  // seg ready; Ah dead -> union region reusable

  // ---- Phase 2: scores = seg @ seg^T (64x64; 16 tiles, 2 per wave)
#pragma unroll
  for (int tt = 0; tt < 2; ++tt) {
    const int tile = wave + 8 * tt;
    const int m0 = (tile >> 2) * 16;
    const int n0 = (tile & 3) * 16;
    v8f acc = {};
#pragma unroll
    for (int ks = 0; ks < 4; ++ks) {
      v16h a  = ldA(sm.segN, m0, ks * 32, SN_STR, lane);
      v16h bf = ldB(sm.segN, n0, ks * 32, SN_STR, lane);  // B^T == seg
      acc = wmma16(a, bf, acc);
    }
#pragma unroll
    for (int v = 0; v < 8; ++v) {
      const int row = m0 + v + 8 * half_id;
      sm.u.p2.scores[row * SC_STR + n0 + l16] = acc[v];
    }
  }
  __syncthreads();

  // ---- Phase 3: row softmax in f32 (64 rows, one thread each)
  if (t < SEG) {
    const float* sr = sm.u.p2.scores + t * SC_STR;
    float mx = -3.0e38f;
    for (int c = 0; c < SEG; ++c) mx = fmaxf(mx, sr[c]);
    float s = 0.f;
    for (int c = 0; c < SEG; ++c) s += __expf(sr[c] - mx);
    const float inv = 1.f / s;
    for (int c = 0; c < SEG; ++c)
      sm.u.p2.attn[t * AT_STR + c] = (_Float16)(__expf(sr[c] - mx) * inv);
  }
  __syncthreads();

  // ---- Phase 4: ctx = attn @ seg -> global out (f32)
  {
    const int m0 = (wave >> 1) * 16;
    v16h af[2];
    af[0] = ldA(sm.u.p2.attn, m0, 0,  AT_STR, lane);
    af[1] = ldA(sm.u.p2.attn, m0, 32, AT_STR, lane);
    float* orow = out + (size_t)start * D_OUT;
#pragma unroll
    for (int nt = 0; nt < 4; ++nt) {
      const int n0 = ((wave & 1) * 4 + nt) * 16;
      v8f acc = {};
#pragma unroll
      for (int ks = 0; ks < 2; ++ks) {
        v16h bf = ldB(sm.segT, n0, ks * 32, ST_STR, lane);  // B^T == segT
        acc = wmma16(af[ks], bf, acc);
      }
      const int col = n0 + l16;
#pragma unroll
      for (int v = 0; v < 8; ++v) {
        const int row = m0 + v + 8 * half_id;
        orow[(size_t)row * D_OUT + col] = acc[v];
      }
    }
  }
}

extern "C" void kernel_launch(void* const* d_in, const int* in_sizes, int n_in,
                              void* d_out, int out_size, void* d_ws, size_t ws_size,
                              hipStream_t stream) {
  (void)n_in; (void)out_size; (void)d_ws; (void)ws_size;
  const float*     h   = (const float*)d_in[0];
  const float*     W   = (const float*)d_in[1];
  const float*     b   = (const float*)d_in[2];
  const long long* sse = (const long long*)d_in[3];
  float* out = (float*)d_out;
  const int nseq = in_sizes[3] / 2;  // 2048 segments
  attn_hidden_fused<<<dim3(nseq), dim3(256), 0, stream>>>(h, W, b, sse, out);
}
